// NetVLAD_21869973471568
// MI455X (gfx1250) — matrix operand order repaired and checked by reading, not compiled, not run on hardware
//
#include <hip/hip_runtime.h>
#include <hip/hip_bf16.h>
#include <math.h>

// Problem constants (from reference)
#define NN 32
#define CC 128
#define LL 4096           // H*W
#define KK 64
#define SPLITS 8          // workgroups per image
#define LCHUNK (LL / SPLITS)  // 512 locations per WG
#define LT 64             // locations per tile
#define NTILES (LCHUNK / LT)  // 8

typedef __attribute__((ext_vector_type(16))) __bf16 v16bf;
typedef __attribute__((ext_vector_type(8)))  float  v8f;

// Fragment unions: word v of the union maps to VGPR v of the WMMA operand,
// matching the ISA 16-bit A/B packing (two bf16 per dword, low = first K).
union FragAB { v16bf v; unsigned u[8]; };
union FragC  { v8f   v; float    f[8]; };

__device__ __forceinline__ unsigned f2bf(float f) {
  unsigned u = __float_as_uint(f);
  u += 0x7FFFu + ((u >> 16) & 1u);   // round-to-nearest-even
  return u >> 16;
}

// CDNA5 async global->LDS copy (ASYNCcnt-tracked, no VGPR data round trip).
__device__ __forceinline__ void async_ld_b128(unsigned lds_off, const void* gaddr) {
  asm volatile("global_load_async_to_lds_b128 %0, %1, off"
               :: "v"(lds_off), "v"(gaddr) : "memory");
}
__device__ __forceinline__ void wait_async0() {
  asm volatile("s_wait_asynccnt 0" ::: "memory");
}

__global__ __launch_bounds__(256)
void netvlad_zero_ws(float* __restrict__ p, int n) {
  int i = blockIdx.x * blockDim.x + threadIdx.x;
  if (i < n) p[i] = 0.f;
}

// Fused: async-staged load -> per-location L2-norm -> GEMM1 (logits) -> softmax
//        -> GEMM2 (vlad accumulate in VGPRs), double-phase pipelined.
__global__ __launch_bounds__(256)
void netvlad_accum(const float* __restrict__ x,
                   const float* __restrict__ conv_w,
                   const float* __restrict__ conv_b,
                   float* __restrict__ vlad_ws,
                   float* __restrict__ asum_ws)
{
  __shared__ __align__(16) float xraw[CC][LT];  // async staging, f32 tile
  __shared__ unsigned short xfT[LT][CC];   // xf[l][c]  (GEMM1 B: c-pairs adjacent)
  __shared__ unsigned short xf [CC][LT];   // xf[c][l]  (GEMM2 B: l-pairs adjacent)
  __shared__ float          logits[KK * LT];
  __shared__ unsigned short a_lds[KK * LT]; // softmax weights, bf16, [k][l]
  __shared__ float          asum_lds[KK];
  __shared__ float          sq_lds[4][LT];
  __shared__ float          rnorm_lds[LT];

  const int tid = threadIdx.x;
  const int w   = tid >> 5;        // wave id 0..7
  const int ln  = tid & 31;        // lane in wave
  const int lhi = ln >> 4;         // 0/1: lane half
  const int llo = ln & 15;

  const int n       = blockIdx.x / SPLITS;
  const int chunk   = blockIdx.x % SPLITS;
  const int lchunk0 = chunk * LCHUNK;

  const int mt  = w & 3;           // row tile (k-cluster /16) for both GEMMs
  const int ntg = w >> 2;          // 0/1
  const int nt0 = ntg * 2;         // GEMM1 column tiles nt0, nt0+1  (l /16)
  const int ct0 = ntg * 4;         // GEMM2 column tiles ct0..ct0+3  (c /16)

  // Async-copy issue geometry: thread covers rows c=(tid>>4)+16p, cols l4=(tid&15)*4
  const int arow = tid >> 4;           // 0..15
  const int acol = (tid & 15) * 4;     // 0,4,..,60
  const unsigned lds_base = (unsigned)(size_t)&xraw[0][0];

  // ---- preload conv_w A-fragments (loop invariant), f32 -> bf16 ----
  FragAB wf[4];
  #pragma unroll
  for (int kk = 0; kk < 4; ++kk) {
    #pragma unroll
    for (int v = 0; v < 8; ++v) {
      int c0 = 32 * kk + ((v >= 4) ? 16 : 0) + 8 * lhi + 2 * (v & 3);
      int kr = 16 * mt + llo;
      unsigned lo = f2bf(conv_w[kr * CC + c0]);
      unsigned hi = f2bf(conv_w[kr * CC + c0 + 1]);
      wf[kk].u[v] = lo | (hi << 16);
    }
  }
  float bias[8];
  #pragma unroll
  for (int j = 0; j < 8; ++j) bias[j] = conv_b[16 * mt + j + 8 * lhi];

  if (tid < KK) asum_lds[tid] = 0.f;

  FragC vacc[4];
  #pragma unroll
  for (int t = 0; t < 4; ++t)
    #pragma unroll
    for (int j = 0; j < 8; ++j) vacc[t].f[j] = 0.f;

  const int lload = tid & 63;   // location within tile (norm phases)
  const int cgrp  = tid >> 6;   // 0..3 -> channels cgrp*32 .. +31

  // ---- prologue: kick off async copy of tile 0 ----
  {
    const int lbase = lchunk0;
    #pragma unroll
    for (int p = 0; p < 8; ++p) {
      int c = arow + 16 * p;
      async_ld_b128(lds_base + (unsigned)((c * LT + acol) * 4),
                    &x[(size_t)(n * CC + c) * LL + lbase + acol]);
    }
  }

  for (int tile = 0; tile < NTILES; ++tile) {
    wait_async0();        // this wave's staged tile is in LDS
    __syncthreads();      // all waves' async writes visible; protects LDS reuse

    // ---- per-location sum of squares from staged tile ----
    {
      float ss = 0.f;
      #pragma unroll
      for (int i = 0; i < 32; ++i)
        ss += xraw[cgrp * 32 + i][lload] * xraw[cgrp * 32 + i][lload];
      sq_lds[cgrp][lload] = ss;
    }
    __syncthreads();
    if (tid < LT) {
      float tot = sq_lds[0][tid] + sq_lds[1][tid] + sq_lds[2][tid] + sq_lds[3][tid];
      rnorm_lds[tid] = 1.f / fmaxf(sqrtf(tot), 1e-12f);
    }
    __syncthreads();
    {
      float rn = rnorm_lds[lload];
      #pragma unroll
      for (int i = 0; i < 32; ++i) {
        int c = cgrp * 32 + i;
        unsigned short b = (unsigned short)f2bf(xraw[c][lload] * rn);
        xf[c][lload]  = b;
        xfT[lload][c] = b;
      }
    }
    __syncthreads();      // xf/xfT ready; all xraw reads complete

    // ---- issue async copy of next tile; overlaps with GEMM/softmax below ----
    if (tile + 1 < NTILES) {
      const int lbase = lchunk0 + (tile + 1) * LT;
      #pragma unroll
      for (int p = 0; p < 8; ++p) {
        int c = arow + 16 * p;
        async_ld_b128(lds_base + (unsigned)((c * LT + acol) * 4),
                      &x[(size_t)(n * CC + c) * LL + lbase + acol]);
      }
    }

    // ---- GEMM1: logits[64,64] = conv_w[64,128] @ xf[128,64] ----
    #pragma unroll
    for (int tnt = 0; tnt < 2; ++tnt) {
      int nt = nt0 + tnt;
      FragC acc;
      #pragma unroll
      for (int j = 0; j < 8; ++j) acc.f[j] = 0.f;
      #pragma unroll
      for (int kk = 0; kk < 4; ++kk) {
        FragAB b;
        #pragma unroll
        for (int v = 0; v < 8; ++v) {
          int l = 16 * nt + llo;
          int c = 32 * kk + ((v >= 4) ? 16 : 0) + 8 * lhi + 2 * (v & 3);
          b.u[v] = *(const unsigned*)&xfT[l][c];
        }
        acc.v = __builtin_amdgcn_wmma_f32_16x16x32_bf16(
            false, wf[kk].v, false, b.v, (short)0, acc.v, false, false);
      }
      #pragma unroll
      for (int j = 0; j < 8; ++j) {
        int kr = 16 * mt + j + 8 * lhi;
        int l  = 16 * nt + llo;
        logits[kr * LT + l] = acc.f[j] + bias[j];
      }
    }
    __syncthreads();

    // ---- softmax over k per location (threads 0..63 own one l each) ----
    if (tid < LT) {
      int l = tid;
      float m = -3.402823466e38f;
      for (int k = 0; k < KK; ++k) m = fmaxf(m, logits[k * LT + l]);
      float s = 0.f;
      for (int k = 0; k < KK; ++k) {
        float e = __expf(logits[k * LT + l] - m);
        logits[k * LT + l] = e;
        s += e;
      }
      float rs = 1.f / s;
      for (int k = 0; k < KK; ++k) {
        float a = logits[k * LT + l] * rs;
        logits[k * LT + l] = a;                       // keep f32 copy for asum
        a_lds[k * LT + l] = (unsigned short)f2bf(a);  // bf16 for WMMA
      }
    }
    __syncthreads();
    if (tid < KK) {   // per-cluster soft-assignment mass
      float s = 0.f;
      for (int l = 0; l < LT; ++l) s += logits[tid * LT + l];
      asum_lds[tid] += s;
    }
    __syncthreads();

    // ---- GEMM2: vlad[64,128] += a[64,64] @ xf^T[64,128] ----
    #pragma unroll
    for (int ks = 0; ks < 2; ++ks) {
      FragAB a;
      #pragma unroll
      for (int v = 0; v < 8; ++v) {
        int kr = 16 * mt + llo;
        int l  = 32 * ks + ((v >= 4) ? 16 : 0) + 8 * lhi + 2 * (v & 3);
        a.u[v] = *(const unsigned*)&a_lds[kr * LT + l];
      }
      #pragma unroll
      for (int t = 0; t < 4; ++t) {
        int ct = ct0 + t;
        FragAB b;
        #pragma unroll
        for (int v = 0; v < 8; ++v) {
          int c = 16 * ct + llo;
          int l = 32 * ks + ((v >= 4) ? 16 : 0) + 8 * lhi + 2 * (v & 3);
          b.u[v] = *(const unsigned*)&xf[c][l];
        }
        vacc[t].v = __builtin_amdgcn_wmma_f32_16x16x32_bf16(
            false, a.v, false, b.v, (short)0, vacc[t].v, false, false);
      }
    }
  }

  // ---- flush partial VLAD + assignment mass to workspace ----
  #pragma unroll
  for (int t = 0; t < 4; ++t) {
    int ct = ct0 + t;
    #pragma unroll
    for (int j = 0; j < 8; ++j) {
      int kr = 16 * mt + j + 8 * lhi;
      int c  = 16 * ct + llo;
      atomicAdd(&vlad_ws[(size_t)(n * KK + kr) * CC + c], vacc[t].f[j]);
    }
  }
  if (tid < KK) atomicAdd(&asum_ws[n * KK + tid], asum_lds[tid]);
}

// centroid correction + intra-normalization + global L2 norm
__global__ __launch_bounds__(256)
void netvlad_finalize(const float* __restrict__ vlad_ws,
                      const float* __restrict__ asum_ws,
                      const float* __restrict__ centroids,
                      float* __restrict__ out)
{
  __shared__ float vbuf[KK * CC];
  __shared__ float red[KK];
  __shared__ float gscale_s;
  const int n = blockIdx.x;
  const int tid = threadIdx.x;

  if (tid < KK) {
    int k = tid;
    float as = asum_ws[n * KK + k];
    float ssq = 0.f;
    for (int c = 0; c < CC; ++c) {
      float v = vlad_ws[(size_t)(n * KK + k) * CC + c] - centroids[k * CC + c] * as;
      vbuf[k * CC + c] = v;
      ssq += v * v;
    }
    float rn = 1.f / fmaxf(sqrtf(ssq), 1e-12f);
    float g = 0.f;
    for (int c = 0; c < CC; ++c) {
      float v = vbuf[k * CC + c] * rn;
      vbuf[k * CC + c] = v;
      g += v * v;
    }
    red[k] = g;
  }
  __syncthreads();
  if (tid == 0) {
    float tot = 0.f;
    for (int k = 0; k < KK; ++k) tot += red[k];
    gscale_s = 1.f / fmaxf(sqrtf(tot), 1e-12f);
  }
  __syncthreads();
  float gs = gscale_s;
  for (int idx = tid; idx < KK * CC; idx += 256)
    out[(size_t)n * KK * CC + idx] = vbuf[idx] * gs;
}

extern "C" void kernel_launch(void* const* d_in, const int* in_sizes, int n_in,
                              void* d_out, int out_size, void* d_ws, size_t ws_size,
                              hipStream_t stream) {
  (void)in_sizes; (void)n_in; (void)out_size; (void)ws_size;
  const float* x      = (const float*)d_in[0];
  const float* conv_w = (const float*)d_in[1];
  const float* conv_b = (const float*)d_in[2];
  const float* cent   = (const float*)d_in[3];
  float* out = (float*)d_out;

  float* vlad_ws = (float*)d_ws;                         // [N,K,C] f32
  float* asum_ws = vlad_ws + (size_t)NN * KK * CC;       // [N,K]   f32

  int ztot = NN * KK * CC + NN * KK;
  hipLaunchKernelGGL(netvlad_zero_ws, dim3((ztot + 255) / 256), dim3(256), 0, stream,
                     vlad_ws, ztot);
  hipLaunchKernelGGL(netvlad_accum, dim3(NN * SPLITS), dim3(256), 0, stream,
                     x, conv_w, conv_b, vlad_ws, asum_ws);
  hipLaunchKernelGGL(netvlad_finalize, dim3(NN), dim3(256), 0, stream,
                     vlad_ws, asum_ws, cent, out);
}
